// GeneralAttention_85899345920686
// MI455X (gfx1250) — matrix operand order repaired and checked by reading, not compiled, not run on hardware
//
#include <hip/hip_runtime.h>
#include <hip/hip_bf16.h>

// Problem constants (reference: B,Q,K,D = 16,2048,2048,512)
#define NB 16
#define NQ 2048
#define NK 2048
#define ND 512

typedef __attribute__((ext_vector_type(16))) _Float16 v16h;
typedef __attribute__((ext_vector_type(8)))  _Float16 v8h;
typedef __attribute__((ext_vector_type(8)))  float    v8f;
typedef __attribute__((ext_vector_type(4)))  float    v4f;

// ---- WMMA wrapper: D = A(16x32 f16) x B(32x16 f16) + C(16x16 f32) ----
static __device__ __forceinline__ v8f wmma16(v16h a, v16h b, v8f c) {
  return __builtin_amdgcn_wmma_f32_16x16x32_f16(
      /*neg_a=*/false, a, /*neg_b=*/false, b,
      /*c_mod=*/(short)0, c, /*reuse_a=*/false, /*reuse_b=*/false);
}

// A fragment from f32 row-major source. p points at row base + kk + ko,
// where ko = 8*(lane>=16). halves[0..7]=K+0..7, halves[8..15]=K+16..23.
static __device__ __forceinline__ v16h a_frag_f32(const float* p) {
  v4f x0 = *(const v4f*)(p);
  v4f x1 = *(const v4f*)(p + 4);
  v4f x2 = *(const v4f*)(p + 16);
  v4f x3 = *(const v4f*)(p + 20);
  v16h r;
#pragma unroll
  for (int j = 0; j < 4; ++j) {
    r[j]      = (_Float16)x0[j];
    r[j + 4]  = (_Float16)x1[j];
    r[j + 8]  = (_Float16)x2[j];
    r[j + 12] = (_Float16)x3[j];
  }
  return r;
}

// A fragment from f16 row-major source (same lane mapping).
static __device__ __forceinline__ v16h a_frag_f16(const _Float16* p) {
  v8h x0 = *(const v8h*)(p);
  v8h x1 = *(const v8h*)(p + 16);
  v16h r;
#pragma unroll
  for (int j = 0; j < 8; ++j) { r[j] = x0[j]; r[j + 8] = x1[j]; }
  return r;
}

// B fragment for C = A * S^T where S is row-major: lane holds K-elements
// base+0..7 and base+16..23 of row (col0 + lane&15), base = kk + 8*(lane>=16).
static __device__ __forceinline__ v16h bt_frag_f16(const _Float16* p) {
  v8h x0 = *(const v8h*)(p);
  v8h x1 = *(const v8h*)(p + 16);
  v16h r;
#pragma unroll
  for (int j = 0; j < 8; ++j) { r[j] = x0[j]; r[j + 8] = x1[j]; }
  return r;
}

// Async global->LDS 16B copy (per lane) + wait, via CDNA5 inline asm.
static __device__ __forceinline__ void async_copy_b128(unsigned lds_off,
                                                       const _Float16* gptr) {
  asm volatile("global_load_async_to_lds_b128 %0, %1, off"
               :
               : "v"(lds_off), "v"(gptr)
               : "memory");
}
static __device__ __forceinline__ void wait_async0() {
  asm volatile("s_wait_asynccnt 0x0" ::: "memory");
}

// ---------------------------------------------------------------------------
// Convert f32 -> f16, 8 elements per thread (exact grids, no tail).
// ---------------------------------------------------------------------------
__global__ void __launch_bounds__(256) k_cvt(const float* __restrict__ in,
                                             _Float16* __restrict__ out) {
  const size_t i = (size_t)blockIdx.x * 256 + threadIdx.x;
  v4f x0 = ((const v4f*)in)[2 * i];
  v4f x1 = ((const v4f*)in)[2 * i + 1];
  v8h r;
#pragma unroll
  for (int j = 0; j < 4; ++j) {
    r[j]     = (_Float16)x0[j];
    r[j + 4] = (_Float16)x1[j];
  }
  ((v8h*)out)[i] = r;
}

// ---------------------------------------------------------------------------
// GEMM1: keys[r,e] = sum_d values[r,d] * W[e,d]   (r = b*NK+k flattened)
// v_f16 [NB*NK, ND], W_f16 [ND, ND], keys f16 [NB*NK, ND]
// Block tile 128x64, wave tile 16x64 (4 WMMA accs), K-loop step 32.
// Inner loop is pure f16 b128 loads + v_wmma.
// ---------------------------------------------------------------------------
__global__ void __launch_bounds__(256) k_keys(const _Float16* __restrict__ vh,
                                              const _Float16* __restrict__ Wh,
                                              _Float16* __restrict__ keys) {
  const int lane = threadIdx.x & 31;
  const int wave = threadIdx.x >> 5;
  const int m    = lane & 15;
  const int kb   = lane >> 4;            // 0 or 1
  const int row0 = blockIdx.x * 128 + wave * 16;
  const int col0 = blockIdx.y * 64;

  v8f acc[4] = {};
  const _Float16* arow = vh + (size_t)(row0 + m) * ND + kb * 8;
  for (int kk = 0; kk < ND; kk += 32) {
    v16h a = a_frag_f16(arow + kk);
    __builtin_prefetch(arow + kk + 32, 0, 3);
#pragma unroll
    for (int t = 0; t < 4; ++t) {
      const _Float16* pb = Wh + (size_t)(col0 + t * 16 + m) * ND + kk + kb * 8;
      acc[t] = wmma16(a, bt_frag_f16(pb), acc[t]);
    }
  }
#pragma unroll
  for (int t = 0; t < 4; ++t)
#pragma unroll
    for (int v = 0; v < 8; ++v)
      keys[(size_t)(row0 + v + 8 * kb) * ND + col0 + t * 16 + m] =
          (_Float16)acc[t][v];
}

// ---------------------------------------------------------------------------
// GEMM2: scores[b][q,k] = sum_d query[b][q,d] * keys[b][k,d]
// q_f16, keys f16 (ws), scores f32 -> attention slice of d_out
// ---------------------------------------------------------------------------
__global__ void __launch_bounds__(256) k_scores(const _Float16* __restrict__ qh,
                                                const _Float16* __restrict__ keys,
                                                float* __restrict__ scores) {
  const int lane = threadIdx.x & 31;
  const int wave = threadIdx.x >> 5;
  const int m    = lane & 15;
  const int kb   = lane >> 4;
  const int b    = blockIdx.z;
  const int row0 = blockIdx.x * 128 + wave * 16;   // q
  const int col0 = blockIdx.y * 64;                // k

  const _Float16* qb  = qh + (size_t)b * NQ * ND;
  const _Float16* kbp = keys + (size_t)b * NK * ND;

  v8f acc[4] = {};
  const _Float16* arow = qb + (size_t)(row0 + m) * ND + kb * 8;
  for (int kk = 0; kk < ND; kk += 32) {
    v16h a = a_frag_f16(arow + kk);
    __builtin_prefetch(arow + kk + 32, 0, 3);
#pragma unroll
    for (int t = 0; t < 4; ++t) {
      const _Float16* pb = kbp + (size_t)(col0 + t * 16 + m) * ND + kk + kb * 8;
      acc[t] = wmma16(a, bt_frag_f16(pb), acc[t]);
    }
  }
  float* out = scores + (size_t)b * NQ * NK;
#pragma unroll
  for (int t = 0; t < 4; ++t)
#pragma unroll
    for (int v = 0; v < 8; ++v)
      out[(size_t)(row0 + v + 8 * kb) * NK + col0 + t * 16 + m] = acc[t][v];
}

// ---------------------------------------------------------------------------
// Softmax over last dim (2048), in place. One block (256 thr) per row.
// ---------------------------------------------------------------------------
__global__ void __launch_bounds__(256) k_softmax(float* __restrict__ att) {
  float* p = att + (size_t)blockIdx.x * NK;
  const int t = threadIdx.x;
  const int lane = t & 31, wave = t >> 5;
  __shared__ float rmax[8];
  __shared__ float rsum[8];

  float x[8];
#pragma unroll
  for (int j = 0; j < 8; ++j) x[j] = p[j * 256 + t];

  float mx = -3.4e38f;
#pragma unroll
  for (int j = 0; j < 8; ++j) mx = fmaxf(mx, x[j]);
#pragma unroll
  for (int o = 16; o > 0; o >>= 1) mx = fmaxf(mx, __shfl_xor(mx, o, 32));
  if (lane == 0) rmax[wave] = mx;
  __syncthreads();
  mx = rmax[0];
#pragma unroll
  for (int w = 1; w < 8; ++w) mx = fmaxf(mx, rmax[w]);

  float s = 0.f;
#pragma unroll
  for (int j = 0; j < 8; ++j) { x[j] = __expf(x[j] - mx); s += x[j]; }
#pragma unroll
  for (int o = 16; o > 0; o >>= 1) s += __shfl_xor(s, o, 32);
  if (lane == 0) rsum[wave] = s;
  __syncthreads();
  s = rsum[0];
#pragma unroll
  for (int w = 1; w < 8; ++w) s += rsum[w];

  const float inv = 1.0f / s;
#pragma unroll
  for (int j = 0; j < 8; ++j) p[j * 256 + t] = x[j] * inv;
}

// ---------------------------------------------------------------------------
// GEMM3: context[b][q,d] = sum_k att[b][q,k] * values[b][k,d]
// B operand needs columns of values: double-buffered async global->LDS
// staging of 32x64 f16 tiles (4KB each), one barrier per k-step.
// ---------------------------------------------------------------------------
__global__ void __launch_bounds__(256) k_context(const float* __restrict__ att,
                                                 const _Float16* __restrict__ vh,
                                                 float* __restrict__ ctx) {
  __shared__ _Float16 Bs[2][32 * 64];
  const int lane = threadIdx.x & 31;
  const int wave = threadIdx.x >> 5;
  const int m    = lane & 15;
  const int kb   = lane >> 4;
  const int b    = blockIdx.z;
  const int row0 = blockIdx.x * 128 + wave * 16;   // q
  const int col0 = blockIdx.y * 64;                // d

  const float*    ab = att + (size_t)b * NQ * NK;
  const _Float16* vb = vh + (size_t)b * NK * ND;

  // staging coords: each thread async-copies 16B (8 halfs) of the 32x64 tile
  const int skr = threadIdx.x >> 3;          // 0..31 (K row within tile)
  const int sns = (threadIdx.x & 7) * 8;     // 0..56 (col within tile)
  const unsigned lds_off[2] = {
      (unsigned)(size_t)&Bs[0][skr * 64 + sns],
      (unsigned)(size_t)&Bs[1][skr * 64 + sns]};
  const _Float16* gsrc = vb + (size_t)skr * ND + col0 + sns;

  // prologue: stage tile for kk = 0 into buffer 0
  async_copy_b128(lds_off[0], gsrc);

  v8f acc[4] = {};
  const float* arow = ab + (size_t)(row0 + m) * NK + kb * 8;
  for (int i = 0; i < NK / 32; ++i) {
    const int kk = i * 32;
    wait_async0();          // our async writes for buffer i&1 landed
    __syncthreads();        // everyone's writes for buffer i&1 landed
    if (kk + 32 < NK)       // stage next tile into the other buffer
      async_copy_b128(lds_off[(i + 1) & 1], gsrc + (size_t)(kk + 32) * ND);

    v16h a = a_frag_f32(arow + kk);
    __builtin_prefetch(arow + kk + 32, 0, 3);
    const _Float16* bs = Bs[i & 1];
#pragma unroll
    for (int t = 0; t < 4; ++t) {
      v16h bfrag;
#pragma unroll
      for (int j = 0; j < 16; ++j)
        bfrag[j] = bs[(kb * 8 + (j & 7) + (j >> 3) * 16) * 64 + t * 16 + m];
      acc[t] = wmma16(a, bfrag, acc[t]);
    }
  }
  float* out = ctx + (size_t)b * NQ * ND;
#pragma unroll
  for (int t = 0; t < 4; ++t)
#pragma unroll
    for (int v = 0; v < 8; ++v)
      out[(size_t)(row0 + v + 8 * kb) * ND + col0 + t * 16 + m] = acc[t][v];
}

// ---------------------------------------------------------------------------
extern "C" void kernel_launch(void* const* d_in, const int* in_sizes, int n_in,
                              void* d_out, int out_size, void* d_ws, size_t ws_size,
                              hipStream_t stream) {
  (void)in_sizes; (void)n_in; (void)out_size; (void)ws_size;
  const float* query  = (const float*)d_in[0];   // [16,2048,512]
  const float* values = (const float*)d_in[1];   // [16,2048,512]
  const float* W      = (const float*)d_in[2];   // [512,512]

  float* ctx_out = (float*)d_out;                              // [16,2048,512]
  float* att_out = ctx_out + (size_t)NB * NQ * ND;             // [16,2048,2048]

  _Float16* v_f16    = (_Float16*)d_ws;                        // 33.5 MB
  _Float16* keys_f16 = v_f16 + (size_t)NB * NK * ND;           // 33.5 MB
  _Float16* q_f16    = keys_f16 + (size_t)NB * NK * ND;        // 33.5 MB
  _Float16* W_f16    = q_f16 + (size_t)NB * NQ * ND;           // 0.5 MB

  // 0) one-shot f32 -> f16 conversions (O(N^2), keeps GEMM loops pure)
  k_cvt<<<dim3((NB * NK * ND / 8) / 256), 256, 0, stream>>>(values, v_f16);
  k_cvt<<<dim3((NB * NQ * ND / 8) / 256), 256, 0, stream>>>(query, q_f16);
  k_cvt<<<dim3((ND * ND / 8) / 256), 256, 0, stream>>>(W, W_f16);

  // 1) keys = values @ W^T  (f16 output into ws)
  {
    dim3 grid((NB * NK) / 128, ND / 64, 1);
    k_keys<<<grid, 256, 0, stream>>>(v_f16, W_f16, keys_f16);
  }
  // 2) scores = query @ keys^T  (f32 into attention slice of d_out)
  {
    dim3 grid(NQ / 128, NK / 64, NB);
    k_scores<<<grid, 256, 0, stream>>>(q_f16, keys_f16, att_out);
  }
  // 3) softmax in place over rows of 2048
  {
    dim3 grid(NB * NQ, 1, 1);
    k_softmax<<<grid, 256, 0, stream>>>(att_out);
  }
  // 4) context = att @ values  (async double-buffered LDS staging)
  {
    dim3 grid(NQ / 128, ND / 64, NB);
    k_context<<<grid, 256, 0, stream>>>(att_out, v_f16, ctx_out);
  }
}
